// AugmentedGeneEmbedding_31808527794912
// MI455X (gfx1250) — matrix operand order repaired and checked by reading, not compiled, not run on hardware
//
#include <hip/hip_runtime.h>
#include <hip/hip_bf16.h>
#include <math.h>

// ---------------------------------------------------------------------------
// Types for WMMA fragments (wave32, gfx1250)
// ---------------------------------------------------------------------------
typedef __attribute__((ext_vector_type(16))) __bf16 v16bf;
typedef __attribute__((ext_vector_type(8)))  __bf16 v8bf;
typedef __attribute__((ext_vector_type(4)))  __bf16 v4bf;
typedef __attribute__((ext_vector_type(8)))  float  v8f;

#define ID_DIM  128
#define ESM_DIM 1280
#define PROJ    256
#define HCAT    384          // ID_DIM + PROJ

#define KT_P 40              // ESM_DIM / 32
#define KT_1 12              // HCAT   / 32
#define KT_2 8               // PROJ   / 32
#define NT_P 16              // PROJ   / 16
#define NT_1 16              // PROJ   / 16
#define NT_2 8               // ID_DIM / 16
#define FRAG_ELEMS 512       // 32 lanes * 16 bf16 per 16x16x32 fragment

#define FRAGS_WP (NT_P * KT_P)   // 640
#define FRAGS_W1 (NT_1 * KT_1)   // 192
#define FRAGS_W2 (NT_2 * KT_2)   // 64

#define M_TILE   32          // tokens per workgroup (2 WMMA m-tiles)
#define THREADS  128         // 4 waves

#define CHUNK_K  128         // esm K streamed per chunk
#define NCHUNK   (ESM_DIM / CHUNK_K)   // 10
#define CHUNK_KT (CHUNK_K / 32)        // 4

#define EBUF_STRIDE 136      // 128 + 8 pad  (row stride % 64 dwords == 4)
#define H_STRIDE    392      // 384 + 8 pad
#define H1_STRIDE   264      // 256 + 8 pad

__device__ __forceinline__ v16bf cat16(v8bf lo, v8bf hi) {
  v16bf r;
#pragma unroll
  for (int i = 0; i < 8; i++) { r[i] = lo[i]; r[i + 8] = hi[i]; }
  return r;
}

__device__ __forceinline__ v8f zero8() {
  v8f z;
#pragma unroll
  for (int i = 0; i < 8; i++) z[i] = 0.0f;
  return z;
}

// B fragment from pre-swizzled global: lane's 16 bf16 are contiguous.
__device__ __forceinline__ v16bf load_b_frag(const __bf16* p) {
  v8bf lo = *(const v8bf*)(p);
  v8bf hi = *(const v8bf*)(p + 8);
  return cat16(lo, hi);
}

// A fragment from row-major bf16 LDS tile. `p` points at row[m] + h*8 + kbase.
// elems 0..7  = K(kbase + h*8 .. +7), elems 8..15 = K(kbase+16 + h*8 .. +7)
__device__ __forceinline__ v16bf load_a_frag(const __bf16* p) {
  v8bf lo = *(const v8bf*)(p);
  v8bf hi = *(const v8bf*)(p + 16);
  return cat16(lo, hi);
}

__device__ __forceinline__ v4bf cvt4(float4 v) {
  v4bf b;
  b[0] = (__bf16)v.x; b[1] = (__bf16)v.y;
  b[2] = (__bf16)v.z; b[3] = (__bf16)v.w;
  return b;
}

#define WMMA_BF16(a, b, c) \
  __builtin_amdgcn_wmma_f32_16x16x32_bf16(false, (a), false, (b), (short)0, (c), false, false)

// ---------------------------------------------------------------------------
// Prep: convert fp32 weights to bf16 in fragment-swizzled layout.
// One block (32 threads = 1 wave) per fragment. Lane l holds:
//   n  = ntBase + (l % 16)
//   kb = ktBase + (l/16)*8 ;  elements e = W[kb+e+(e>=8?8:0)][n]
// ---------------------------------------------------------------------------
__global__ __launch_bounds__(32)
void prep_weights(const float* __restrict__ Wp, const float* __restrict__ W1,
                  const float* __restrict__ W2,
                  __bf16* __restrict__ WpS, __bf16* __restrict__ W1S,
                  __bf16* __restrict__ W2S) {
  int f = blockIdx.x;
  int lane = threadIdx.x;

  const float* W; __bf16* S; int N, nKT, fi;
  if (f < FRAGS_WP)                { W = Wp; S = WpS; N = PROJ;   nKT = KT_P; fi = f; }
  else if (f < FRAGS_WP + FRAGS_W1){ W = W1; S = W1S; N = PROJ;   nKT = KT_1; fi = f - FRAGS_WP; }
  else                             { W = W2; S = W2S; N = ID_DIM; nKT = KT_2; fi = f - FRAGS_WP - FRAGS_W1; }

  int nt = fi / nKT;
  int kt = fi % nKT;
  int n  = nt * 16 + (lane & 15);
  int kb = kt * 32 + (lane >> 4) * 8;

  __bf16* dst = S + (size_t)fi * FRAG_ELEMS + (size_t)lane * 16;
#pragma unroll
  for (int e = 0; e < 16; e++)
    dst[e] = (__bf16)W[(size_t)(kb + e + (e >= 8 ? 8 : 0)) * N + n];
}

// ---------------------------------------------------------------------------
// Fused kernel: 32 tokens per workgroup, 128 threads (4 waves).
// Each wave computes 2 m-tiles x 4 n-tiles -> every B fragment is reused for
// two WMMAs (halves L2 weight traffic vs a 16-token tile).
//   stage 0: token meta + id rows -> LDS bf16
//   stage 1: proj = esm @ Wp (double-buffered K=128 chunks), +bp, mask,
//            *tanh(gate) -> lds_h[:,128:384]
//   stage 2: h1 = gelu(h @ W1 + b1) -> lds_h1
//   stage 3: out = h1 @ W2 + b2 -> global fp32 (non-temporal)
// ---------------------------------------------------------------------------
__global__ __launch_bounds__(THREADS, 1)
void fused_gene_embed(const int* __restrict__ idx,
                      const int* __restrict__ g2e,
                      const float* __restrict__ id_table,
                      const float* __restrict__ esm_table,
                      const float* __restrict__ bp,
                      const float* __restrict__ gate,
                      const float* __restrict__ b1,
                      const float* __restrict__ b2,
                      const __bf16* __restrict__ WpS,
                      const __bf16* __restrict__ W1S,
                      const __bf16* __restrict__ W2S,
                      float* __restrict__ out,
                      int TOK, int n_genes, int v_esm) {
  __shared__ __align__(16) __bf16 lds_ebuf[2][M_TILE][EBUF_STRIDE]; // 17408 B
  __shared__ __align__(16) __bf16 lds_h[M_TILE][H_STRIDE];          // 25088 B
  __shared__ __align__(16) __bf16 lds_h1[M_TILE][H1_STRIDE];        // 16896 B
  __shared__ int   s_gene[M_TILE];
  __shared__ int   s_erow[M_TILE];
  __shared__ float s_scale[M_TILE];

  const int tid  = threadIdx.x;
  const int lane = tid & 31;
  const int wave = tid >> 5;
  const int tileBase = blockIdx.x * M_TILE;

  // ---- stage 0a: token metadata -------------------------------------------
  if (tid < M_TILE) {
    int tok = tileBase + tid;
    if (tok >= TOK) tok = TOK - 1;
    int g  = idx[tok];
    int gc = g < 0 ? 0 : (g >= n_genes ? n_genes - 1 : g);
    int e  = g2e[gc];
    bool valid = (g >= 0) && (g < n_genes) && (e > 0) && (e < v_esm);
    s_gene[tid]  = gc;
    s_erow[tid]  = valid ? e : 0;
    s_scale[tid] = valid ? tanhf(gate[0]) : 0.0f;
  }
  __syncthreads();

  // staging role: 4 threads per row
  const int row = tid >> 2;   // 0..31
  const int q   = tid & 3;    // 0..3
  const float* esrc = esm_table + (size_t)s_erow[row] * ESM_DIM;

  // ---- stage 0b: id rows -> lds_h[:,0:128]; esm chunk 0 -> ebuf[0] --------
  {
    const float* isrc = id_table + (size_t)s_gene[row] * ID_DIM;
#pragma unroll
    for (int i = 0; i < 8; i++) {
      int c = (q + i * 4) * 4;                 // 0..124
      float4 v = *(const float4*)(isrc + c);
      *(v4bf*)&lds_h[row][c] = cvt4(v);
    }
#pragma unroll
    for (int i = 0; i < 8; i++) {
      int c = (q + i * 4) * 4;                 // 0..124
      float4 v = *(const float4*)(esrc + c);
      *(v4bf*)&lds_ebuf[0][row][c] = cvt4(v);
    }
  }
  __syncthreads();

  const int m = lane & 15;   // A row (within m-tile) / B-C column
  const int h = lane >> 4;   // lane half

  // ---- stage 1: proj = esm @ Wp (double-buffered chunks) ------------------
  {
    v8f acc[2][4];
#pragma unroll
    for (int mt = 0; mt < 2; mt++)
#pragma unroll
      for (int j = 0; j < 4; j++) acc[mt][j] = zero8();

    const __bf16* arow0 = &lds_ebuf[0][m][h * 8];
    const __bf16* arow1 = &lds_ebuf[0][m + 16][h * 8];
    const size_t  bufStride = (size_t)M_TILE * EBUF_STRIDE; // elems per buffer
    const __bf16* bbase = WpS + (size_t)lane * 16;

    for (int c = 0; c < NCHUNK; c++) {
      int p = c & 1;
      // prefetch next chunk into registers (overlaps with WMMAs below)
      float4 pf[8];
      if (c + 1 < NCHUNK) {
        const float* src = esrc + (c + 1) * CHUNK_K;
#pragma unroll
        for (int i = 0; i < 8; i++)
          pf[i] = *(const float4*)(src + (q + i * 4) * 4);
      }
      // 4 WMMA k-steps on current chunk
#pragma unroll
      for (int kk = 0; kk < CHUNK_KT; kk++) {
        int kt = c * CHUNK_KT + kk;
        v16bf a0 = load_a_frag(arow0 + p * bufStride + kk * 32);
        v16bf a1 = load_a_frag(arow1 + p * bufStride + kk * 32);
#pragma unroll
        for (int j = 0; j < 4; j++) {
          int nt = wave * 4 + j;
          v16bf b = load_b_frag(bbase + (size_t)(nt * KT_P + kt) * FRAG_ELEMS);
          acc[0][j] = WMMA_BF16(a0, b, acc[0][j]);
          acc[1][j] = WMMA_BF16(a1, b, acc[1][j]);
        }
      }
      // commit prefetched chunk into the other buffer
      if (c + 1 < NCHUNK) {
#pragma unroll
        for (int i = 0; i < 8; i++)
          *(v4bf*)&lds_ebuf[p ^ 1][row][(q + i * 4) * 4] = cvt4(pf[i]);
      }
      __syncthreads();
    }

    // epilogue: +bp, mask * tanh(gate), write bf16 to concat buffer
#pragma unroll
    for (int j = 0; j < 4; j++) {
      int nb = (wave * 4 + j) * 16;
      float bpv = bp[nb + m];
#pragma unroll
      for (int mt = 0; mt < 2; mt++) {
#pragma unroll
        for (int r = 0; r < 8; r++) {
          int mr = mt * 16 + h * 8 + r;
          float val = (acc[mt][j][r] + bpv) * s_scale[mr];
          lds_h[mr][ID_DIM + nb + m] = (__bf16)val;
        }
      }
    }
  }
  __syncthreads();

  // ---- stage 2: h1 = gelu(h @ W1 + b1) ------------------------------------
  {
    v8f acc[2][4];
#pragma unroll
    for (int mt = 0; mt < 2; mt++)
#pragma unroll
      for (int j = 0; j < 4; j++) acc[mt][j] = zero8();

    const __bf16* arow0 = &lds_h[m][h * 8];
    const __bf16* arow1 = &lds_h[m + 16][h * 8];
    const __bf16* bbase = W1S + (size_t)lane * 16;
    for (int kt = 0; kt < KT_1; kt++) {
      v16bf a0 = load_a_frag(arow0 + kt * 32);
      v16bf a1 = load_a_frag(arow1 + kt * 32);
#pragma unroll
      for (int j = 0; j < 4; j++) {
        int nt = wave * 4 + j;
        v16bf b = load_b_frag(bbase + (size_t)(nt * KT_1 + kt) * FRAG_ELEMS);
        acc[0][j] = WMMA_BF16(a0, b, acc[0][j]);
        acc[1][j] = WMMA_BF16(a1, b, acc[1][j]);
      }
    }
#pragma unroll
    for (int j = 0; j < 4; j++) {
      int nb = (wave * 4 + j) * 16;
      float b1v = b1[nb + m];
#pragma unroll
      for (int mt = 0; mt < 2; mt++) {
#pragma unroll
        for (int r = 0; r < 8; r++) {
          int mr = mt * 16 + h * 8 + r;
          float x = acc[mt][j][r] + b1v;
          float g = 0.5f * x * (1.0f + erff(x * 0.70710678118654752f));
          lds_h1[mr][nb + m] = (__bf16)g;
        }
      }
    }
  }
  __syncthreads();

  // ---- stage 3: out = h1 @ W2 + b2 ----------------------------------------
  {
    v8f acc[2][2];
#pragma unroll
    for (int mt = 0; mt < 2; mt++)
#pragma unroll
      for (int j = 0; j < 2; j++) acc[mt][j] = zero8();

    const __bf16* arow0 = &lds_h1[m][h * 8];
    const __bf16* arow1 = &lds_h1[m + 16][h * 8];
    const __bf16* bbase = W2S + (size_t)lane * 16;
    for (int kt = 0; kt < KT_2; kt++) {
      v16bf a0 = load_a_frag(arow0 + kt * 32);
      v16bf a1 = load_a_frag(arow1 + kt * 32);
#pragma unroll
      for (int j = 0; j < 2; j++) {
        int nt = wave * 2 + j;
        v16bf b = load_b_frag(bbase + (size_t)(nt * KT_2 + kt) * FRAG_ELEMS);
        acc[0][j] = WMMA_BF16(a0, b, acc[0][j]);
        acc[1][j] = WMMA_BF16(a1, b, acc[1][j]);
      }
    }
#pragma unroll
    for (int j = 0; j < 2; j++) {
      int nb = (wave * 2 + j) * 16;
      float b2v = b2[nb + m];
#pragma unroll
      for (int mt = 0; mt < 2; mt++) {
#pragma unroll
        for (int r = 0; r < 8; r++) {
          int mr = mt * 16 + h * 8 + r;
          int tok = tileBase + mr;
          if (tok < TOK)
            __builtin_nontemporal_store(acc[mt][j][r] + b2v,
                                        &out[(size_t)tok * ID_DIM + nb + m]);
        }
      }
    }
  }
}

// ---------------------------------------------------------------------------
// Host launcher
// ---------------------------------------------------------------------------
extern "C" void kernel_launch(void* const* d_in, const int* in_sizes, int n_in,
                              void* d_out, int out_size, void* d_ws, size_t ws_size,
                              hipStream_t stream) {
  const int*   idx       = (const int*)d_in[0];
  const int*   g2e       = (const int*)d_in[1];
  const float* id_table  = (const float*)d_in[2];
  const float* esm_table = (const float*)d_in[3];
  const float* Wp        = (const float*)d_in[4];
  const float* bp        = (const float*)d_in[5];
  const float* gate      = (const float*)d_in[6];
  const float* W1        = (const float*)d_in[7];
  const float* b1        = (const float*)d_in[8];
  const float* W2        = (const float*)d_in[9];
  const float* b2        = (const float*)d_in[10];
  float*       out       = (float*)d_out;

  int TOK     = in_sizes[0];
  int n_genes = in_sizes[1];
  int v_esm   = in_sizes[3] / ESM_DIM;

  // workspace layout: swizzled bf16 weights (~0.9 MB total)
  __bf16* WpS = (__bf16*)d_ws;
  __bf16* W1S = WpS + (size_t)FRAGS_WP * FRAG_ELEMS;
  __bf16* W2S = W1S + (size_t)FRAGS_W1 * FRAG_ELEMS;

  prep_weights<<<FRAGS_WP + FRAGS_W1 + FRAGS_W2, 32, 0, stream>>>(
      Wp, W1, W2, WpS, W1S, W2S);

  int nTiles = (TOK + M_TILE - 1) / M_TILE;
  fused_gene_embed<<<nTiles, THREADS, 0, stream>>>(
      idx, g2e, id_table, esm_table, bp, gate, b1, b2,
      WpS, W1S, W2S, out, TOK, n_genes, v_esm);
}
